// BidirectionalCrossAttention_75419625718428
// MI455X (gfx1250) — matrix operand order repaired and checked by reading, not compiled, not run on hardware
//
#include <hip/hip_runtime.h>
#include <hip/hip_bf16.h>

#define DD 512
#define BB 8
#define SS 2048
#define MM (BB * SS)
#define KBLK 32

typedef __attribute__((ext_vector_type(2)))  float  v2f;
typedef __attribute__((ext_vector_type(8)))  float  v8f;
typedef __attribute__((ext_vector_type(16))) __bf16 v16bf;
typedef unsigned short u16b;

static __device__ __forceinline__ u16b f32_to_bf16u(float f) {
    union { float f; unsigned u; } c; c.f = f;
    unsigned r = c.u + 0x7fffu + ((c.u >> 16) & 1u);  // round-to-nearest-even
    return (u16b)(r >> 16);
}
static __device__ __forceinline__ __bf16 bf_from_bits(u16b x) {
    union { u16b u; __bf16 b; } c; c.u = x; return c.b;
}
static __device__ __forceinline__ __bf16 f32_to_bf16(float f) {
    return bf_from_bits(f32_to_bf16u(f));
}

// 16-bit A-matrix 16x32 layout (contiguous, matches B fills): VGPR g holds K pair;
// lanes 0-15 get K in {0..15}, lanes 16-31 get K in {16..31}.
static __device__ __forceinline__ int a_koff(int g, int hf) {
    return hf * 16 + 2 * g;
}

// 16B async global->LDS copy, tracked by ASYNCcnt (CDNA5 ISA)
static __device__ __forceinline__ void async_copy16(unsigned lds_off, const void* gaddr) {
    asm volatile("global_load_async_to_lds_b128 %0, %1, off"
                 :: "v"(lds_off), "v"(gaddr) : "memory");
}

// ---------------------------------------------------------------------------
// FP32 GEMM + bias via V_WMMA_F32_16X16X4_F32 :  H[M,512] = X[M,K] @ W[K,512] + b
// Block: 256 threads (8 waves); wave w computes a 16x64 tile; block covers 16x512.
// ---------------------------------------------------------------------------
__global__ __launch_bounds__(256) void proj_f32_kernel(
    const float* __restrict__ X, const float* __restrict__ W,
    const float* __restrict__ bias, float* __restrict__ H, int K)
{
    const int lane = threadIdx.x & 31;
    const int wave = threadIdx.x >> 5;
    const int hf   = lane >> 4;
    const int mm   = lane & 15;
    const int m0   = blockIdx.x * 16;
    const int n0   = wave * 64;

    v8f acc[4] = {};
    const float* xrow = X + (size_t)(m0 + mm) * K;

    for (int kb = 0; kb < K; kb += 4) {
        // 32-bit A 16x4: lanes0-15 -> K=0(v0),1(v1); lanes16-31 -> K=2,3
        v2f a;
        a.x = xrow[kb + 2 * hf + 0];
        a.y = xrow[kb + 2 * hf + 1];
        #pragma unroll
        for (int t = 0; t < 4; ++t) {
            int n = n0 + t * 16 + mm;
            v2f bm;
            bm.x = W[(size_t)(kb + 2 * hf + 0) * DD + n];
            bm.y = W[(size_t)(kb + 2 * hf + 1) * DD + n];
            acc[t] = __builtin_amdgcn_wmma_f32_16x16x4_f32(
                false, a, false, bm, (short)0, acc[t], false, false);
        }
    }
    #pragma unroll
    for (int t = 0; t < 4; ++t) {
        int n = n0 + t * 16 + mm;
        #pragma unroll
        for (int j = 0; j < 8; ++j) {
            int m = m0 + j + 8 * hf;  // C layout: VGPR j -> row j (+8 for hi half)
            H[(size_t)m * DD + n] = acc[t][j] + bias[n];
        }
    }
}

// ---------------------------------------------------------------------------
// BF16 GEMM + bias via V_WMMA_F32_16X16X32_BF16: P = H[M,512] @ W[512,512] + b
// W chunk [32,512] is staged to LDS as bf16 once per block (coalesced loads,
// one convert per element) instead of per-wave strided gathers.
// transpose==0 : outN[m*512+n]  (row-major bf16, Q and K)
// transpose==1 : outT[(batch*512+n)*2048 + s]  (d-major bf16, V -> P@V B tiles)
// ---------------------------------------------------------------------------
__global__ __launch_bounds__(256) void proj_qkv_kernel(
    const float* __restrict__ H, const float* __restrict__ W,
    const float* __restrict__ bias, u16b* __restrict__ outN,
    u16b* __restrict__ outT, int transpose)
{
    __shared__ u16b Wbuf[32 * DD];  // 32KB: current 32-row bf16 slice of W

    const int lane = threadIdx.x & 31;
    const int wave = threadIdx.x >> 5;
    const int hf   = lane >> 4;
    const int mm   = lane & 15;
    const int m0   = blockIdx.x * 16;
    const int n0   = wave * 64;

    v8f acc[4] = {};
    const float* hrow = H + (size_t)(m0 + mm) * DD;

    for (int kb = 0; kb < DD; kb += 32) {
        __syncthreads();  // previous chunk's Wbuf reads complete
        for (int c = 0; c < 64; ++c) {
            int idx = c * 256 + threadIdx.x;  // 32*512 elems, coalesced
            Wbuf[idx] = f32_to_bf16u(W[(size_t)kb * DD + idx]);
        }
        __syncthreads();

        v16bf a;
        #pragma unroll
        for (int g = 0; g < 8; ++g) {
            int k = kb + a_koff(g, hf);
            a[2 * g]     = f32_to_bf16(hrow[k]);
            a[2 * g + 1] = f32_to_bf16(hrow[k + 1]);
        }
        #pragma unroll
        for (int t = 0; t < 4; ++t) {
            int n = n0 + t * 16 + mm;
            v16bf bm;  // B 32x16: lanes0-15 rows 0..15, lanes16-31 rows 16..31
            #pragma unroll
            for (int g = 0; g < 8; ++g) {
                int k = hf * 16 + 2 * g;
                bm[2 * g]     = bf_from_bits(Wbuf[k * DD + n]);
                bm[2 * g + 1] = bf_from_bits(Wbuf[(k + 1) * DD + n]);
            }
            acc[t] = __builtin_amdgcn_wmma_f32_16x16x32_bf16(
                false, a, false, bm, (short)0, acc[t], false, false);
        }
    }
    #pragma unroll
    for (int t = 0; t < 4; ++t) {
        int n = n0 + t * 16 + mm;
        #pragma unroll
        for (int j = 0; j < 8; ++j) {
            int m = m0 + j + 8 * hf;
            u16b v = f32_to_bf16u(acc[t][j] + bias[n]);
            if (!transpose) {
                outN[(size_t)m * DD + n] = v;
            } else {
                int batch = m >> 11, s = m & 2047;
                outT[((size_t)(batch * DD + n)) * SS + s] = v;
            }
        }
    }
}

// ---------------------------------------------------------------------------
// Flash attention: O = softmax(Q K^T / sqrt(512)) V, accumulated into Fused.
// Block: 128 threads = 4 waves; block owns (batch, 16 queries); wave w owns
// d-chunk [128w,128w+128). K/V key-blocks are double-buffered in LDS via
// GLOBAL_LOAD_ASYNC_TO_LDS_B128 (ASYNCcnt), prefetching block i+1 while the
// WMMAs consume block i. Partial scores reduced across waves via LDS.
// ---------------------------------------------------------------------------
__global__ __launch_bounds__(128) void flash_attn_kernel(
    const u16b* __restrict__ Q, const u16b* __restrict__ Kk,
    const u16b* __restrict__ Vt, float* __restrict__ Fused, int accumulate)
{
    __shared__ u16b  Kbuf[2][KBLK * DD];   // 2 x 32KB: K block [32 keys][512 d]
    __shared__ u16b  Vbuf[2][DD][KBLK];    // 2 x 32KB: V block [512 d][32 keys]
    __shared__ float ldsS[4][2][16][16];   // per-wave partial score tiles
    __shared__ u16b  ldsP[4][16][32];      // wave-private P C->A layout staging

    const int tid  = threadIdx.x;
    const int lane = tid & 31;
    const int wave = tid >> 5;
    const int hf   = lane >> 4;
    const int mm   = lane & 15;
    const int batch = blockIdx.x >> 7;          // 128 query-blocks per batch
    const int q0    = (blockIdx.x & 127) * 16;
    const int d0    = wave * 128;
    const float scale = 0.044194173824159216f;  // 1/sqrt(512)

    const u16b* Kbase = Kk + (size_t)batch * SS * DD;  // [2048][512]
    const u16b* Vbase = Vt + (size_t)batch * DD * SS;  // [512][2048]

    // Issue one key-block prefetch: 32 async b128 ops per wave (16 K + 16 V).
    auto prefetch = [&](int blk, int buf) {
        const u16b* kg = Kbase + (size_t)blk * KBLK * DD;  // contiguous 32KB
        unsigned kl = (unsigned)(size_t)&Kbuf[buf][0];
        #pragma unroll
        for (int c = 0; c < 16; ++c) {
            int off = (c * 128 + tid) * 8;  // u16 units, 16B chunks
            async_copy16(kl + off * 2, kg + off);
        }
        unsigned vl = (unsigned)(size_t)&Vbuf[buf][0][0];
        #pragma unroll
        for (int c = 0; c < 16; ++c) {
            int ci = c * 128 + tid;          // 512 rows x 4 x 16B
            int row = ci >> 2, part = ci & 3;
            async_copy16(vl + (row * KBLK + part * 8) * 2,
                         Vbase + (size_t)row * SS + blk * KBLK + part * 8);
        }
    };

    prefetch(0, 0);

    // Q A-tiles for this wave's d-chunk: 4 tiles of 16x32 bf16
    v16bf qa[4];
    const u16b* qrow = Q + ((size_t)(batch * SS + q0 + mm)) * DD + d0;
    #pragma unroll
    for (int kt = 0; kt < 4; ++kt) {
        #pragma unroll
        for (int g = 0; g < 8; ++g) {
            int k = kt * 32 + a_koff(g, hf);
            qa[kt][2 * g]     = bf_from_bits(qrow[k]);
            qa[kt][2 * g + 1] = bf_from_bits(qrow[k + 1]);
        }
    }

    v8f o[8] = {};
    float rm[8], rs[8];
    #pragma unroll
    for (int j = 0; j < 8; ++j) { rm[j] = -3.0e38f; rs[j] = 0.0f; }

    for (int i = 0; i < SS / KBLK; ++i) {
        const int cur = i & 1;
        if (i + 1 < SS / KBLK) {
            prefetch(i + 1, cur ^ 1);
            asm volatile("s_wait_asynccnt 0x20" ::: "memory");  // block i done
        } else {
            asm volatile("s_wait_asynccnt 0x0" ::: "memory");
        }
        __syncthreads();  // all waves' async copies for block i visible

        // ---- partial scores S[16q,32k] over this wave's d-chunk ----
        v8f st[2] = {};
        #pragma unroll
        for (int t2 = 0; t2 < 2; ++t2) {
            const u16b* krow = &Kbuf[cur][(t2 * 16 + mm) * DD + d0 + hf * 16];
            #pragma unroll
            for (int kt = 0; kt < 4; ++kt) {
                v16bf kv;  // B-tile: lane=key column, d rows contiguous
                #pragma unroll
                for (int g = 0; g < 8; ++g) {
                    kv[2 * g]     = bf_from_bits(krow[kt * 32 + 2 * g]);
                    kv[2 * g + 1] = bf_from_bits(krow[kt * 32 + 2 * g + 1]);
                }
                st[t2] = __builtin_amdgcn_wmma_f32_16x16x32_bf16(
                    false, qa[kt], false, kv, (short)0, st[t2], false, false);
            }
        }

        // ---- reduce partial scores across the 4 waves via LDS ----
        #pragma unroll
        for (int t2 = 0; t2 < 2; ++t2)
            #pragma unroll
            for (int j = 0; j < 8; ++j)
                ldsS[wave][t2][j + 8 * hf][mm] = st[t2][j];
        __syncthreads();
        #pragma unroll
        for (int t2 = 0; t2 < 2; ++t2)
            #pragma unroll
            for (int j = 0; j < 8; ++j) {
                int r = j + 8 * hf;
                st[t2][j] = (ldsS[0][t2][r][mm] + ldsS[1][t2][r][mm] +
                             ldsS[2][t2][r][mm] + ldsS[3][t2][r][mm]) * scale;
            }

        // ---- online softmax (row stats via 16-lane xor-shuffle reduce) ----
        float corr[8];
        #pragma unroll
        for (int j = 0; j < 8; ++j) {
            float mx = fmaxf(st[0][j], st[1][j]);
            #pragma unroll
            for (int d = 1; d <= 8; d <<= 1) mx = fmaxf(mx, __shfl_xor(mx, d, 32));
            float mnew = fmaxf(rm[j], mx);
            corr[j] = __expf(rm[j] - mnew);
            float p0 = __expf(st[0][j] - mnew);
            float p1 = __expf(st[1][j] - mnew);
            st[0][j] = p0; st[1][j] = p1;
            float sm = p0 + p1;
            #pragma unroll
            for (int d = 1; d <= 8; d <<= 1) sm += __shfl_xor(sm, d, 32);
            rs[j] = rs[j] * corr[j] + sm;
            rm[j] = mnew;
        }
        #pragma unroll
        for (int ot = 0; ot < 8; ++ot)
            #pragma unroll
            for (int j = 0; j < 8; ++j)
                o[ot][j] *= corr[j];

        // ---- P: C-layout -> bf16 A-layout via wave-private LDS ----
        #pragma unroll
        for (int t2 = 0; t2 < 2; ++t2)
            #pragma unroll
            for (int j = 0; j < 8; ++j)
                ldsP[wave][j + 8 * hf][t2 * 16 + mm] = f32_to_bf16u(st[t2][j]);
        asm volatile("s_wait_dscnt 0" ::: "memory");

        v16bf pa;
        #pragma unroll
        for (int g = 0; g < 8; ++g) {
            int k = a_koff(g, hf);
            pa[2 * g]     = bf_from_bits(ldsP[wave][mm][k]);
            pa[2 * g + 1] = bf_from_bits(ldsP[wave][mm][k + 1]);
        }

        // ---- O += P @ V  (V block d-major in LDS -> contiguous B-tiles) ----
        #pragma unroll
        for (int ot = 0; ot < 8; ++ot) {
            const u16b* vrow = &Vbuf[cur][d0 + ot * 16 + mm][hf * 16];
            v16bf vv;
            #pragma unroll
            for (int g = 0; g < 8; ++g) {
                vv[2 * g]     = bf_from_bits(vrow[2 * g]);
                vv[2 * g + 1] = bf_from_bits(vrow[2 * g + 1]);
            }
            o[ot] = __builtin_amdgcn_wmma_f32_16x16x32_bf16(
                false, pa, false, vv, (short)0, o[ot], false, false);
        }

        __syncthreads();  // all reads of buf[cur]/ldsS done before next prefetch
    }

    #pragma unroll
    for (int j = 0; j < 8; ++j) rs[j] = 1.0f / rs[j];
    #pragma unroll
    for (int ot = 0; ot < 8; ++ot) {
        int n = d0 + ot * 16 + mm;
        #pragma unroll
        for (int j = 0; j < 8; ++j) {
            size_t idx = ((size_t)(batch * SS + q0 + j + 8 * hf)) * DD + n;
            float v = o[ot][j] * rs[j];
            if (accumulate) Fused[idx] += v; else Fused[idx] = v;
        }
    }
}

// ---------------------------------------------------------------------------
// out = gamma * fused + h1 + h2
// ---------------------------------------------------------------------------
__global__ __launch_bounds__(256) void final_kernel(
    const float* __restrict__ fused, const float* __restrict__ h1,
    const float* __restrict__ h2, const float* __restrict__ gamma,
    float* __restrict__ out, int n)
{
    int i = (blockIdx.x * 256 + threadIdx.x) * 4;
    if (i >= n) return;
    float g = gamma[0];
    float4 f = *(const float4*)(fused + i);
    float4 a = *(const float4*)(h1 + i);
    float4 b = *(const float4*)(h2 + i);
    float4 r;
    r.x = g * f.x + a.x + b.x;
    r.y = g * f.y + a.y + b.y;
    r.z = g * f.z + a.z + b.z;
    r.w = g * f.w + a.w + b.w;
    *(float4*)(out + i) = r;
}

extern "C" void kernel_launch(void* const* d_in, const int* in_sizes, int n_in,
                              void* d_out, int out_size, void* d_ws, size_t ws_size,
                              hipStream_t stream) {
    const float* x1  = (const float*)d_in[0];
    const float* x2  = (const float*)d_in[1];
    const float* Wp1 = (const float*)d_in[2];
    const float* bp1 = (const float*)d_in[3];
    const float* Wp2 = (const float*)d_in[4];
    const float* bp2 = (const float*)d_in[5];
    const float* Wq  = (const float*)d_in[6];
    const float* bq  = (const float*)d_in[7];
    const float* Wk  = (const float*)d_in[8];
    const float* bk  = (const float*)d_in[9];
    const float* Wv  = (const float*)d_in[10];
    const float* bv  = (const float*)d_in[11];
    const float* gamma = (const float*)d_in[12];
    float* out = (float*)d_out;

    char* ws = (char*)d_ws;
    const size_t szF = (size_t)MM * DD * sizeof(float);   // 33.5 MB
    const size_t szH = (size_t)MM * DD * sizeof(u16b);    // 16.7 MB
    float* h1    = (float*)ws; ws += szF;
    float* h2    = (float*)ws; ws += szF;
    float* fused = (float*)ws; ws += szF;
    u16b*  qB    = (u16b*)ws;  ws += szH;
    u16b*  kB    = (u16b*)ws;  ws += szH;
    u16b*  vT    = (u16b*)ws;  ws += szH;

    // fp32 projections (dominate output since gamma == 0)
    proj_f32_kernel<<<dim3(MM / 16), dim3(256), 0, stream>>>(x1, Wp1, bp1, h1, 768);
    proj_f32_kernel<<<dim3(MM / 16), dim3(256), 0, stream>>>(x2, Wp2, bp2, h2, 1024);

    // direction 1: queries from h1, keys/values from h2
    proj_qkv_kernel<<<dim3(MM / 16), dim3(256), 0, stream>>>(h1, Wq, bq, qB, (u16b*)nullptr, 0);
    proj_qkv_kernel<<<dim3(MM / 16), dim3(256), 0, stream>>>(h2, Wk, bk, kB, (u16b*)nullptr, 0);
    proj_qkv_kernel<<<dim3(MM / 16), dim3(256), 0, stream>>>(h2, Wv, bv, (u16b*)nullptr, vT, 1);
    flash_attn_kernel<<<dim3(BB * SS / 16), dim3(128), 0, stream>>>(qB, kB, vT, fused, 0);

    // direction 2: queries from h2, keys/values from h1
    proj_qkv_kernel<<<dim3(MM / 16), dim3(256), 0, stream>>>(h2, Wq, bq, qB, (u16b*)nullptr, 0);
    proj_qkv_kernel<<<dim3(MM / 16), dim3(256), 0, stream>>>(h1, Wk, bk, kB, (u16b*)nullptr, 0);
    proj_qkv_kernel<<<dim3(MM / 16), dim3(256), 0, stream>>>(h1, Wv, bv, (u16b*)nullptr, vT, 1);
    flash_attn_kernel<<<dim3(BB * SS / 16), dim3(128), 0, stream>>>(qB, kB, vT, fused, 1);

    final_kernel<<<dim3((MM * DD) / (256 * 4)), dim3(256), 0, stream>>>(
        fused, h1, h2, gamma, out, MM * DD);
}
